// FGRM_55147380081058
// MI455X (gfx1250) — matrix operand order repaired and checked by reading, not compiled, not run on hardware
//
#include <hip/hip_runtime.h>
#include <cstdint>

// ---------------------------------------------------------------------------
// CDNA5 (gfx1250) implementation: implicit-GEMM bf16 WMMA convolutions
// (register-blocked 16ch x 32px per wave, K-chunked LDS im2col),
// fused BN-stat reduction, fused elementwise glue kernels.
// ---------------------------------------------------------------------------

typedef __attribute__((ext_vector_type(16))) __bf16 v16bf;
typedef __attribute__((ext_vector_type(8)))  float  v8f;

#define B_   16
#define C1_  64
#define C2_  128
#define H2_  64
#define W2_  64
#define H_   128
#define W_   128

// ---------------------------------------------------------------------------
// small utility kernels
// ---------------------------------------------------------------------------
__global__ void k_zero(float* p, int n) {
    int i = blockIdx.x * blockDim.x + threadIdx.x;
    if (i < n) p[i] = 0.f;
}

// Repack OIHW f32 weights into WMMA A-fragment order (bf16).
// Fragment (mtile, kt) occupies 32 lanes x 16 halves contiguously:
//   idx = ((mtile*KT + kt)*32 + lane)*16 + i
//   M = mtile*16 + (lane&15)
//   K = kt*32 + i + ((lane&16)?16:0)      (contiguous k order, matches B)
template<int Cout, int Cin>
__global__ void k_repack(const float* __restrict__ w, __bf16* __restrict__ wp) {
    constexpr int K = Cin * 9, KT = K / 32;
    int idx = blockIdx.x * blockDim.x + threadIdx.x;
    if (idx >= Cout * K) return;
    int i    = idx & 15;
    int lane = (idx >> 4) & 31;
    int rest = idx >> 9;
    int kt   = rest % KT;
    int mt   = rest / KT;
    int M  = mt * 16 + (lane & 15);
    int Kk = kt * 32 + i + ((lane & 16) ? 16 : 0);
    int cin = Kk / 9, rem = Kk % 9;
    wp[idx] = (__bf16)w[((M * Cin + cin) * 3 + rem / 3) * 3 + (rem % 3)];
}

// imap = sigmoid(bilinear-up x2, align_corners=True)  : [B,1,128,128]
__global__ void k_imap(const float* __restrict__ inmap, float* __restrict__ imap) {
    int idx = blockIdx.x * blockDim.x + threadIdx.x;
    if (idx >= B_ * H_ * W_) return;
    int X = idx & 127, Y = (idx >> 7) & 127, n = idx >> 14;
    float sy = Y * (63.0f / 127.0f), sx = X * (63.0f / 127.0f);
    int y0 = (int)sy, x0 = (int)sx;
    int y1 = min(y0 + 1, 63), x1 = min(x0 + 1, 63);
    float wy = sy - (float)y0, wx = sx - (float)x0;
    const float* p = inmap + (size_t)n * (H2_ * W2_);
    float a = p[y0 * 64 + x0] * (1.f - wy) + p[y1 * 64 + x0] * wy;
    float b = p[y0 * 64 + x1] * (1.f - wy) + p[y1 * 64 + x1] * wy;
    float v = a * (1.f - wx) + b * wx;
    imap[idx] = 1.f / (1.f + __expf(-v));
}

// increase_map = dilate3x3(imap) - imap   (SAME, -inf pad == max over valid)
__global__ void k_incmap(const float* __restrict__ imap, float* __restrict__ inc) {
    int idx = blockIdx.x * blockDim.x + threadIdx.x;
    if (idx >= B_ * H_ * W_) return;
    int X = idx & 127, Y = (idx >> 7) & 127, n = idx >> 14;
    const float* p = imap + (size_t)n * (H_ * W_);
    float c = p[Y * W_ + X];
    float m = c;
    #pragma unroll
    for (int dy = -1; dy <= 1; ++dy) {
        int gy = Y + dy;
        if ((unsigned)gy >= (unsigned)H_) continue;
        #pragma unroll
        for (int dx = -1; dx <= 1; ++dx) {
            int gx = X + dx;
            if ((unsigned)gx >= (unsigned)W_) continue;
            m = fmaxf(m, p[gy * W_ + gx]);
        }
    }
    inc[idx] = m - c;
}

// ---------------------------------------------------------------------------
// Implicit-GEMM 3x3 conv via v_wmma_f32_16x16x32_bf16.
//   GEMM: M=Cout, N=B*H*W pixels, K=Cin*9.
//   Block = 128 threads = 4 waves = 4 mtiles; each wave register-blocks
//   16 channels x 32 pixels (two v8f accumulators) so each A fragment
//   (one 32B load/lane) feeds TWO WMMAs.
//   LDS im2col patch [pixel][Kchunk] staged in 576-wide K chunks (36 KB),
//   so Cin=128 runs as two barrier-separated chunks with the same LDS.
//   HAS_MAP fuses  in * increase_map  into the im2col fill (conv2 path).
//   doStats: per-channel sum/sum^2 via 16-lane shuffle tree + atomics (BN).
// ---------------------------------------------------------------------------
template<int Cin, int Cout, int H, int W, bool HAS_MAP, typename Tin>
__global__ __launch_bounds__(128)
void k_conv3(const Tin* __restrict__ in, const float* __restrict__ map,
             const __bf16* __restrict__ wp, const float* __restrict__ bias,
             float* __restrict__ out, float* __restrict__ ssum,
             float* __restrict__ ssq, int doStats) {
    constexpr int K     = Cin * 9;
    constexpr int KC    = 576;          // K-chunk staged in LDS
    constexpr int KTtot = K / 32;
    constexpr int TILE  = 32;           // pixels per block
    __shared__ __bf16 sm[TILE * KC];    // 36 KB

    const int tid = threadIdx.x;
    const int tilesPerRow = W / TILE;
    const int t  = blockIdx.x;
    const int n  = t / (H * tilesPerRow);
    const int r2 = t - n * (H * tilesPerRow);
    const int y  = r2 / tilesPerRow;
    const int x0 = (r2 - y * tilesPerRow) * TILE;

    const int lane = tid & 31;
    const int wv   = tid >> 5;              // 4 waves = 4 mtiles
    const int mtg  = blockIdx.y * 4 + wv;

    v8f acc0 = {}, acc1 = {};
    const __bf16* bp0 = sm + (size_t)(lane & 15) * KC + ((lane & 16) ? 16 : 0);
    const __bf16* bp1 = bp0 + 16 * KC;

    for (int kc = 0; kc < K; kc += KC) {
        if (kc) __syncthreads();            // uniform: safe
        // --- im2col fill: sm[x*KC + kl] = in[n, cin, y+r-1, x0+x+s-1] (*map)
        for (int idx = tid; idx < TILE * KC; idx += 128) {   // exact multiple
            int x  = idx / KC;
            int kl = idx - x * KC;
            int k  = kc + kl;
            int cin = k / 9;
            int rem = k - cin * 9;
            int r = rem / 3;
            int s = rem - r * 3;
            int gy = y + r - 1;
            int gx = x0 + x + s - 1;
            float v = 0.f;
            if ((unsigned)gy < (unsigned)H && (unsigned)gx < (unsigned)W) {
                v = (float)in[(((size_t)n * Cin + cin) * H + gy) * W + gx];
                if (HAS_MAP) v *= map[((size_t)n * H + gy) * W + gx];
            }
            sm[idx] = (__bf16)v;
        }
        __syncthreads();

        // --- WMMA loop over this K chunk: 1 A load -> 2 WMMAs ---------------
        const __bf16* ap = wp + ((size_t)mtg * KTtot + (kc >> 5)) * 512 + lane * 16;
        #pragma unroll 2
        for (int kt = 0; kt < KC / 32; ++kt) {
            v16bf a  = *(const v16bf*)ap;
            v16bf b0 = *(const v16bf*)(bp0 + kt * 32);
            v16bf b1 = *(const v16bf*)(bp1 + kt * 32);
            __builtin_prefetch(ap + 512, 0, 1);   // global_prefetch next A frag
            ap += 512;
            acc0 = __builtin_amdgcn_wmma_f32_16x16x32_bf16(
                       false, a, false, b0, (short)0, acc0, false, false);
            acc1 = __builtin_amdgcn_wmma_f32_16x16x32_bf16(
                       false, a, false, b1, (short)0, acc1, false, false);
        }
    }

    // --- epilogue: bias, store f32, BN partial stats -------------------------
    const int chB = mtg * 16 + ((lane & 16) ? 8 : 0);
    const int gx0 = x0 + (lane & 15);
    const bool leader = (lane & 15) == 0;
    #pragma unroll
    for (int v = 0; v < 8; ++v) {
        int ch = chB + v;
        float b  = bias[ch];
        float v0 = acc0[v] + b;
        float v1 = acc1[v] + b;
        size_t base = (((size_t)n * Cout + ch) * H + y) * W;
        out[base + gx0]      = v0;
        out[base + gx0 + 16] = v1;
        if (doStats) {
            float s1 = v0 + v1;
            float s2 = v0 * v0 + v1 * v1;
            #pragma unroll
            for (int o = 8; o > 0; o >>= 1) {    // reduce across 16 lanes/half-wave
                s1 += __shfl_xor(s1, o, 16);
                s2 += __shfl_xor(s2, o, 16);
            }
            if (leader) {
                atomicAdd(&ssum[ch], s1);
                atomicAdd(&ssq[ch], s2);
            }
        }
    }
}

// BN finalize: scale = g*rsqrt(var+eps), shift = be - mean*scale
__global__ void k_bnfin(const float* __restrict__ ssum, const float* __restrict__ ssq,
                        const float* __restrict__ g, const float* __restrict__ be,
                        float* __restrict__ scale, float* __restrict__ shift,
                        float invN, int C) {
    int c = blockIdx.x * blockDim.x + threadIdx.x;
    if (c >= C) return;
    float m   = ssum[c] * invN;
    float var = ssq[c] * invN - m * m;
    float sc  = g[c] * rsqrtf(var + 1e-5f);
    scale[c] = sc;
    shift[c] = be[c] - m * sc;
}

// x1 = pixel_shuffle2(relu(bn(up_raw))) + beta * fn   -> bf16
__global__ void k_shuffle_add(const float* __restrict__ upraw,
                              const float* __restrict__ scale, const float* __restrict__ shift,
                              const float* __restrict__ fn, const float* __restrict__ beta,
                              __bf16* __restrict__ x1) {
    int idx = blockIdx.x * blockDim.x + threadIdx.x;
    if (idx >= B_ * C1_ * H_ * W_) return;
    int X = idx & 127, Y = (idx >> 7) & 127, c = (idx >> 14) & 63, n = idx >> 20;
    int h = Y >> 1, i = Y & 1, w = X >> 1, j = X & 1;
    int c4 = c * 4 + i * 2 + j;
    float d = upraw[(((size_t)n * (4 * C1_) + c4) * H2_ + h) * W2_ + w] * scale[c4] + shift[c4];
    d = fmaxf(d, 0.f);
    x1[idx] = (__bf16)(d + beta[0] * fn[idx]);
}

// relu(bn(x)) -> bf16 (+ optional f32 copy into d_out for final r)
template<bool F32OUT>
__global__ void k_bnrelu(const float* __restrict__ raw,
                         const float* __restrict__ scale, const float* __restrict__ shift,
                         __bf16* __restrict__ obf, float* __restrict__ of32) {
    int idx = blockIdx.x * blockDim.x + threadIdx.x;
    if (idx >= B_ * C1_ * H_ * W_) return;
    int c = (idx >> 14) & 63;
    float v = fmaxf(raw[idx] * scale[c] + shift[c], 0.f);
    obf[idx] = (__bf16)v;
    if (F32OUT) of32[idx] = v;
}

// 7x7 conv, 64 -> 1 channel, pad 3 (memory-bound; VALU fp32 accumulate)
__global__ void k_outconv(const __bf16* __restrict__ r, const float* __restrict__ w,
                          const float* __restrict__ b, float* __restrict__ out) {
    int idx = blockIdx.x * blockDim.x + threadIdx.x;
    if (idx >= B_ * H_ * W_) return;
    int X = idx & 127, Y = (idx >> 7) & 127, n = idx >> 14;
    float acc = b[0];
    for (int c = 0; c < C1_; ++c) {
        const __bf16* rp = r + (((size_t)n * C1_ + c) * H_) * W_;
        const float* wp = w + c * 49;
        #pragma unroll
        for (int dy = 0; dy < 7; ++dy) {
            int gy = Y + dy - 3;
            if ((unsigned)gy >= (unsigned)H_) continue;
            #pragma unroll
            for (int dx = 0; dx < 7; ++dx) {
                int gx = X + dx - 3;
                if ((unsigned)gx < (unsigned)W_)
                    acc += (float)rp[gy * W_ + gx] * wp[dy * 7 + dx];
            }
        }
    }
    out[idx] = acc;
}

// ---------------------------------------------------------------------------
// launch
// ---------------------------------------------------------------------------
extern "C" void kernel_launch(void* const* d_in, const int* in_sizes, int n_in,
                              void* d_out, int out_size, void* d_ws, size_t ws_size,
                              hipStream_t stream) {
    const float* cur_x  = (const float*)d_in[0];
    const float* dep_x  = (const float*)d_in[1];
    const float* in_map = (const float*)d_in[2];
    const float* up_w   = (const float*)d_in[3];
    const float* up_b   = (const float*)d_in[4];
    const float* up_g   = (const float*)d_in[5];
    const float* up_be  = (const float*)d_in[6];
    const float* c2_w   = (const float*)d_in[7];
    const float* c2_b   = (const float*)d_in[8];
    const float* beta   = (const float*)d_in[9];
    const float* dw[3]  = { (const float*)d_in[10], (const float*)d_in[14], (const float*)d_in[18] };
    const float* db[3]  = { (const float*)d_in[11], (const float*)d_in[15], (const float*)d_in[19] };
    const float* dg[3]  = { (const float*)d_in[12], (const float*)d_in[16], (const float*)d_in[20] };
    const float* dbe[3] = { (const float*)d_in[13], (const float*)d_in[17], (const float*)d_in[21] };
    const float* out_w  = (const float*)d_in[22];
    const float* out_b  = (const float*)d_in[23];

    float* out_r   = (float*)d_out;                       // [16,64,128,128]
    float* out_map = out_r + (size_t)B_ * C1_ * H_ * W_;  // [16,1,128,128]

    // --- workspace carve ----------------------------------------------------
    char* ws = (char*)d_ws;
    size_t off = 0;
    auto alloc = [&](size_t bytes) -> void* {
        void* p = ws + off;
        off = (off + bytes + 255) & ~(size_t)255;
        return p;
    };
    float*  stats  = (float*)alloc(1792 * sizeof(float));
    float*  ssum_up = stats + 0,   *ssq_up = stats + 256;
    float*  sc_up   = stats + 512, *sh_up  = stats + 768;
    float*  ssum_d[3], *ssq_d[3], *sc_d[3], *sh_d[3];
    for (int i = 0; i < 3; ++i) {
        float* base = stats + 1024 + i * 256;
        ssum_d[i] = base; ssq_d[i] = base + 64; sc_d[i] = base + 128; sh_d[i] = base + 192;
    }
    __bf16* wp_up = (__bf16*)alloc((size_t)256 * 1152 * 2);
    __bf16* wp_c2 = (__bf16*)alloc((size_t)64 * 576 * 2);
    __bf16* wp_d[3];
    for (int i = 0; i < 3; ++i) wp_d[i] = (__bf16*)alloc((size_t)64 * 576 * 2);
    float*  imap = (float*)alloc((size_t)B_ * H_ * W_ * 4);
    float*  inc  = (float*)alloc((size_t)B_ * H_ * W_ * 4);
    float*  R1   = (float*)alloc((size_t)B_ * C1_ * H_ * W_ * 4);       // 67 MB
    float*  R2   = (float*)alloc((size_t)B_ * 4 * C1_ * H2_ * W2_ * 4); // 67 MB
    __bf16* X1   = (__bf16*)alloc((size_t)B_ * C1_ * H_ * W_ * 2);      // 33.5 MB
    __bf16* X2   = (__bf16*)alloc((size_t)B_ * C1_ * H_ * W_ * 2);      // 33.5 MB

    const int EW = (B_ * C1_ * H_ * W_ + 255) / 256;   // 65536 blocks
    const int PM = (B_ * H_ * W_ + 255) / 256;         // 1024 blocks
    const dim3 GD(B_ * H_ * (W_ / 32), 1);             // d-conv grid: 8192 x 1
    const dim3 GU(B_ * H2_ * (W2_ / 32), 4);           // up-conv grid: 2048 x 4

    // 0) zero BN stat accumulators
    k_zero<<<7, 256, 0, stream>>>(stats, 1792);

    // 1) weight repacks to WMMA fragment order
    k_repack<256, 128><<<(256 * 1152 + 255) / 256, 256, 0, stream>>>(up_w, wp_up);
    k_repack<64, 64><<<(64 * 576 + 255) / 256, 256, 0, stream>>>(c2_w, wp_c2);
    for (int i = 0; i < 3; ++i)
        k_repack<64, 64><<<(64 * 576 + 255) / 256, 256, 0, stream>>>(dw[i], wp_d[i]);

    // 2) attention map
    k_imap<<<PM, 256, 0, stream>>>(in_map, imap);
    k_incmap<<<PM, 256, 0, stream>>>(imap, inc);

    // 3) fn = conv2(cur_x * increase_map)   [map fused into im2col fill]
    k_conv3<64, 64, 128, 128, true, float><<<GD, 128, 0, stream>>>(
        cur_x, inc, wp_c2, c2_b, R1, ssum_up, ssq_up, 0);

    // 4) up conv: dep_x (128 -> 256) @64x64, with BN stats (two K chunks)
    k_conv3<128, 256, 64, 64, false, float><<<GU, 128, 0, stream>>>(
        dep_x, inc, wp_up, up_b, R2, ssum_up, ssq_up, 1);
    k_bnfin<<<1, 256, 0, stream>>>(ssum_up, ssq_up, up_g, up_be, sc_up, sh_up,
                                   1.0f / (B_ * H2_ * W2_), 256);

    // 5) x1 = pixel_shuffle(relu(bn(up))) + beta*fn  -> bf16
    k_shuffle_add<<<EW, 256, 0, stream>>>(R2, sc_up, sh_up, R1, beta, X1);

    // 6) d1 / d2 / d3 conv + BN + ReLU chain
    k_conv3<64, 64, 128, 128, false, __bf16><<<GD, 128, 0, stream>>>(
        X1, inc, wp_d[0], db[0], R1, ssum_d[0], ssq_d[0], 1);
    k_bnfin<<<1, 64, 0, stream>>>(ssum_d[0], ssq_d[0], dg[0], dbe[0], sc_d[0], sh_d[0],
                                  1.0f / (B_ * H_ * W_), 64);
    k_bnrelu<false><<<EW, 256, 0, stream>>>(R1, sc_d[0], sh_d[0], X2, out_r);

    k_conv3<64, 64, 128, 128, false, __bf16><<<GD, 128, 0, stream>>>(
        X2, inc, wp_d[1], db[1], R2, ssum_d[1], ssq_d[1], 1);
    k_bnfin<<<1, 64, 0, stream>>>(ssum_d[1], ssq_d[1], dg[1], dbe[1], sc_d[1], sh_d[1],
                                  1.0f / (B_ * H_ * W_), 64);
    k_bnrelu<false><<<EW, 256, 0, stream>>>(R2, sc_d[1], sh_d[1], X1, out_r);

    k_conv3<64, 64, 128, 128, false, __bf16><<<GD, 128, 0, stream>>>(
        X1, inc, wp_d[2], db[2], R1, ssum_d[2], ssq_d[2], 1);
    k_bnfin<<<1, 64, 0, stream>>>(ssum_d[2], ssq_d[2], dg[2], dbe[2], sc_d[2], sh_d[2],
                                  1.0f / (B_ * H_ * W_), 64);
    k_bnrelu<true><<<EW, 256, 0, stream>>>(R1, sc_d[2], sh_d[2], X2, out_r);  // r -> d_out + bf16

    // 7) output_map = conv7x7(r) + bias
    k_outconv<<<PM, 256, 0, stream>>>(X2, out_w, out_b, out_map);
}
